// CustomDistanceLayer_41987600286359
// MI455X (gfx1250) — matrix operand order repaired and checked by reading, not compiled, not run on hardware
//
#include <hip/hip_runtime.h>

// Problem geometry (from reference): K=32 window, H=4096 canvas, OUTW=H-K+1.
#define KW    32
#define H     4096
#define OUTW  4065
#define TILES 255            // ceil(OUTW/16)
#define RS_STRIDE 4080       // TILES*16, multiple of 16, >= OUTW
#define EPS   1e-8f

typedef __attribute__((ext_vector_type(16))) _Float16 v16h;
typedef __attribute__((ext_vector_type(8)))  float    v8f;
typedef __attribute__((ext_vector_type(4)))  float    vf4;

// ---------------------------------------------------------------------------
// Pass A: per-row variance-weighted squared distance + horizontal 32-box-sum.
// One block (256 threads) per row. som/var are stream-once data: read with
// non-temporal hints so they don't evict the L2-resident rs buffer. dist row
// lives in LDS with stride-17 padding (idx = x + (x>>4)) so the 16-stride
// sliding windows are bank-conflict-free across the 64 LDS banks.
// ---------------------------------------------------------------------------
__global__ __launch_bounds__(256) void dist_rowsum_kernel(
    const float* __restrict__ img,   // 32x32
    const float* __restrict__ som,   // 4096x4096
    const float* __restrict__ var,   // 4096x4096
    float* __restrict__ rs)          // 4096 x RS_STRIDE (workspace)
{
  __shared__ float simg[KW];
  __shared__ float sdist[H + (H >> 4)];   // padded dist row

  const int y = blockIdx.x;
  const int t = threadIdx.x;

  if (t < KW) simg[t] = img[(y & (KW - 1)) * KW + t];
  __syncthreads();

  const long rowbase = (long)y * H;
  const int  x0 = t * 16;

  const vf4* somv = (const vf4*)(som + rowbase + x0);   // 16B-aligned
  const vf4* varv = (const vf4*)(var + rowbase + x0);
  #pragma unroll
  for (int i4 = 0; i4 < 4; ++i4) {
    const vf4 sv = __builtin_nontemporal_load(&somv[i4]);
    const vf4 vv = __builtin_nontemporal_load(&varv[i4]);
    #pragma unroll
    for (int cix = 0; cix < 4; ++cix) {
      const int   x = x0 + i4 * 4 + cix;
      const float d = simg[x & (KW - 1)] - sv[cix];
      sdist[x + (x >> 4)] = d * d / (vv[cix] + EPS);
    }
  }
  __syncthreads();

  if (t < TILES) {                      // t==255 -> j0=4080 > 4064: skip
    const int j0 = x0;
    float s = 0.0f;
    #pragma unroll
    for (int i = 0; i < KW; ++i) { const int x = j0 + i; s += sdist[x + (x >> 4)]; }
    float acc[16];
    acc[0] = s;
    #pragma unroll
    for (int q = 1; q < 16; ++q) {
      const int xin  = j0 + KW - 1 + q;
      const int xout = j0 + q - 1;
      s += sdist[xin + (xin >> 4)] - sdist[xout + (xout >> 4)];
      acc[q] = s;
    }
    float* rrow = rs + (long)y * RS_STRIDE;
    if (j0 + 15 < OUTW) {
      #pragma unroll
      for (int q = 0; q < 16; ++q) rrow[j0 + q] = acc[q];   // contiguous 64B/lane
    } else {
      #pragma unroll
      for (int q = 0; q < 16; ++q) if (j0 + q < OUTW) rrow[j0 + q] = acc[q];
    }
  }
}

// ---------------------------------------------------------------------------
// Pass B: vertical 32-box-sum as a banded GEMM on the WMMA unit.
//   out[i0+m][j0+n] = sum_{k=m..m+31} rs[i0+k][j0+n]
// One wave handles a 16-row band (tile_i) and 4 consecutive 16x16 output
// tiles along j, reusing the lane-resident 0/1 band matrix A (f16, exact)
// across 8 v_wmma_f32_16x16x32_f16 ops.
// B loads are unconditional; interior tiles use one base + constant offsets.
// D stores: interior tiles (wave-uniform test) store unguarded through one
// base + constant r*OUTW offsets (no exec-mask churn); only the last tile
// row/col takes the per-lane-guarded path. Output is write-once: store
// non-temporal so it doesn't evict the L2-resident rs buffer.
// Element->(M,N,K) mapping used here (wave32; A and B share this k order):
//   A: M = lane&15, element e -> K = e + 16*half
//   B: N = lane&15, element e -> K = e + 16*half
//   C/D: VGPR r -> row r + 8*half, col = lane&15
// ---------------------------------------------------------------------------
__global__ __launch_bounds__(256) void vsum_wmma_kernel(
    const float* __restrict__ rs,    // 4096 x RS_STRIDE
    float* __restrict__ out)         // OUTW x OUTW
{
  const int wave   = threadIdx.x >> 5;
  const int lane   = threadIdx.x & 31;
  const int tile_i = blockIdx.x * 8 + wave;   // wave-uniform
  if (tile_i >= TILES) return;

  const int i0   = tile_i * 16;
  const int half = lane >> 4;
  const int mn   = lane & 15;      // M for A, N for B/C/D

  // Band matrix A (depends only on lane) — hoisted across all 4 j-tiles.
  v16h a0, a1;
  #pragma unroll
  for (int e = 0; e < 16; ++e) {
    const int ka = e + 16 * half;
    a0[e] = (ka >= mn && ka <= mn + 31) ? (_Float16)1.0f : (_Float16)0.0f;
    const int kb = ka + 32;
    a1[e] = (kb >= mn && kb <= mn + 31) ? (_Float16)1.0f : (_Float16)0.0f;
  }

  #pragma unroll
  for (int jt = 0; jt < 4; ++jt) {
    const int tile_j = blockIdx.y * 4 + jt;
    if (tile_j >= TILES) break;                 // uniform
    const int j0   = tile_j * 16;
    const int jcol = j0 + mn;

    v16h b0, b1;
    if (tile_i < 253) {
      // Interior: rows i0 .. i0+63 all < H. Constant-offset loads.
      const float* base = rs + (long)(i0 + 16 * half) * RS_STRIDE + jcol;
      #pragma unroll
      for (int e = 0; e < 16; ++e) {
        b0[e] = (_Float16)base[(long)e * RS_STRIDE];
        b1[e] = (_Float16)base[(long)(e + 32) * RS_STRIDE];
      }
    } else {
      // Edge: clamp rows into [0, H-1]; clamped garbage only hits A==0
      // terms or unstored output rows.
      #pragma unroll
      for (int e = 0; e < 16; ++e) {
        const int k0 = e + 16 * half;
        const int r0 = min(i0 + k0, H - 1);
        const int r1 = min(i0 + k0 + 32, H - 1);
        b0[e] = (_Float16)rs[(long)r0 * RS_STRIDE + jcol];
        b1[e] = (_Float16)rs[(long)r1 * RS_STRIDE + jcol];
      }
    }

    v8f c = {0.f, 0.f, 0.f, 0.f, 0.f, 0.f, 0.f, 0.f};
    c = __builtin_amdgcn_wmma_f32_16x16x32_f16(false, a0, false, b0,
                                               (short)0, c, false, false);
    c = __builtin_amdgcn_wmma_f32_16x16x32_f16(false, a1, false, b1,
                                               (short)0, c, false, false);

    float* obase = out + (long)(i0 + 8 * half) * OUTW + jcol;
    if (i0 + 15 < OUTW && j0 + 15 < OUTW) {     // wave-uniform: interior tile
      #pragma unroll
      for (int r = 0; r < 8; ++r)
        __builtin_nontemporal_store(c[r], &obase[(long)r * OUTW]);
    } else {                                    // last tile row/col only
      #pragma unroll
      for (int r = 0; r < 8; ++r) {
        const int row = i0 + r + 8 * half;
        if (row < OUTW && jcol < OUTW)
          __builtin_nontemporal_store(c[r], &obase[(long)r * OUTW]);
      }
    }
  }
}

// ---------------------------------------------------------------------------
// Fallback (only if ws_size can't hold the 66.8 MB row-sum buffer): direct
// 32x32 accumulation per output. Slow but correct; keeps launch deterministic.
// ---------------------------------------------------------------------------
__global__ __launch_bounds__(256) void naive_fallback_kernel(
    const float* __restrict__ img,
    const float* __restrict__ som,
    const float* __restrict__ var,
    float* __restrict__ out)
{
  const long idx = (long)blockIdx.x * blockDim.x + threadIdx.x;
  if (idx >= (long)OUTW * OUTW) return;
  const int oy = (int)(idx / OUTW);
  const int ox = (int)(idx % OUTW);
  float s = 0.0f;
  for (int dy = 0; dy < KW; ++dy) {
    const int  row  = oy + dy;
    const long base = (long)row * H;
    const int  irow = (row & (KW - 1)) * KW;
    for (int dx = 0; dx < KW; ++dx) {
      const int   x = ox + dx;
      const float d = img[irow + (x & (KW - 1))] - som[base + x];
      s += d * d / (var[base + x] + EPS);
    }
  }
  out[idx] = s;
}

extern "C" void kernel_launch(void* const* d_in, const int* in_sizes, int n_in,
                              void* d_out, int out_size, void* d_ws, size_t ws_size,
                              hipStream_t stream) {
  const float* img = (const float*)d_in[0];   // 32x32
  const float* som = (const float*)d_in[1];   // 4096x4096
  const float* var = (const float*)d_in[2];   // 4096x4096
  float* out = (float*)d_out;                 // 4065x4065
  float* rs  = (float*)d_ws;

  const size_t rs_bytes = (size_t)H * RS_STRIDE * sizeof(float);
  if (ws_size >= rs_bytes) {
    dist_rowsum_kernel<<<H, 256, 0, stream>>>(img, som, var, rs);
    vsum_wmma_kernel<<<dim3((TILES + 7) / 8, (TILES + 3) / 4), 256, 0, stream>>>(rs, out);
  } else {
    const long n = (long)OUTW * OUTW;
    naive_fallback_kernel<<<(unsigned)((n + 255) / 256), 256, 0, stream>>>(img, som, var, out);
  }
}
